// GatedDeltaNetBlock_80272938762619
// MI455X (gfx1250) — matrix operand order repaired and checked by reading, not compiled, not run on hardware
//
#include <hip/hip_runtime.h>

// ---------------- problem constants (match reference) ----------------
constexpr int kB   = 2;
constexpr int kT   = 1024;
constexpr int kD   = 1024;
constexpr int kH   = 16;
constexpr int kDK  = 64;
constexpr int kDV  = 128;
constexpr int kFH  = 2816;
constexpr int kCONV = 4;
constexpr int kNT  = kB * kT;          // 2048 token rows
constexpr int kCH  = 64;               // chunk length for delta rule
constexpr int kNC  = kT / kCH;         // 16 chunks per sequence
constexpr int kNU  = kB * kH * kNC;    // 512 (b,h,chunk) units

typedef __attribute__((ext_vector_type(16))) __bf16 v16bf;
typedef __attribute__((ext_vector_type(8)))  float  v8f;

union PK8 { __bf16 h[8]; uint4 u; };          // 8 bf16 <-> one 16B access
union PK2 { __bf16 h[2]; unsigned int u; };   // 2 bf16 <-> one 4B access

// ---------------- WMMA GEMM: C[M,N] = A[M,K] * B[K,N], fp32 in/out, bf16 MACs
// Block tile 128(M) x 128(N), K-step 32. 256 threads = 8 waves (4M x 2N),
// each wave owns 32x64 = 2x4 wmma tiles. Double-buffered LDS, 1 barrier/iter.
#define BM 128
#define BN 128
#define BK 32

template <int N>
__global__ __launch_bounds__(256)
void wmma_gemm_kernel(const float* __restrict__ A, const float* __restrict__ Bm,
                      float* __restrict__ C, int M, int K) {
  __shared__ __align__(16) __bf16 As[2][BM * BK];   // [m][k], row stride BK
  __shared__ __align__(16) __bf16 Bs[2][BN * BK];   // [n][k] transposed, row stride BK

  const int tid  = threadIdx.x;
  const int lane = tid & 31;
  const int wid  = tid >> 5;
  const int wm   = wid & 3;        // 0..3 -> m offset wm*32
  const int wn   = wid >> 2;       // 0..1 -> n offset wn*64
  const int l16  = lane & 15;
  const int hlf  = lane >> 4;      // 0/1

  const int m0 = blockIdx.y * BM;
  const int n0 = blockIdx.x * BN;

  const int arow = tid >> 1;            // 0..127
  const int acol = (tid & 1) * 16;      // 0 or 16
  const int bk   = (tid & 15) * 2;      // 0..30 (k pair)
  const int bn   = (tid >> 4) * 8;      // 0..120 (8 n values)

  v8f acc[2][4];
#pragma unroll
  for (int i = 0; i < 2; ++i)
#pragma unroll
    for (int j = 0; j < 4; ++j)
#pragma unroll
      for (int r = 0; r < 8; ++r) acc[i][j][r] = 0.0f;

  float4 aR[4];
  float4 bR[4];

  {
    const float4* ap = reinterpret_cast<const float4*>(A + (size_t)(m0 + arow) * K + acol);
#pragma unroll
    for (int j = 0; j < 4; ++j) aR[j] = ap[j];
    const float4* bp0 = reinterpret_cast<const float4*>(Bm + (size_t)bk * N + n0 + bn);
    const float4* bp1 = reinterpret_cast<const float4*>(Bm + (size_t)(bk + 1) * N + n0 + bn);
    bR[0] = bp0[0]; bR[1] = bp0[1]; bR[2] = bp1[0]; bR[3] = bp1[1];
  }
  {
    PK8 p0, p1;
    const float* af = reinterpret_cast<const float*>(aR);
#pragma unroll
    for (int j = 0; j < 8; ++j) { p0.h[j] = (__bf16)af[j]; p1.h[j] = (__bf16)af[8 + j]; }
    *reinterpret_cast<uint4*>(&As[0][arow * BK + acol])     = p0.u;
    *reinterpret_cast<uint4*>(&As[0][arow * BK + acol + 8]) = p1.u;
    const float* b0 = reinterpret_cast<const float*>(&bR[0]);
    const float* b1 = reinterpret_cast<const float*>(&bR[2]);
#pragma unroll
    for (int j = 0; j < 8; ++j) {
      PK2 p; p.h[0] = (__bf16)b0[j]; p.h[1] = (__bf16)b1[j];
      *reinterpret_cast<unsigned int*>(&Bs[0][(bn + j) * BK + bk]) = p.u;
    }
  }

  int buf = 0;
  for (int k0 = 0; k0 < K; k0 += BK) {
    __syncthreads();

    const bool more = (k0 + BK) < K;
    if (more) {
      const int kn = k0 + BK;
      const float4* ap = reinterpret_cast<const float4*>(A + (size_t)(m0 + arow) * K + kn + acol);
#pragma unroll
      for (int j = 0; j < 4; ++j) aR[j] = ap[j];
      const float4* bp0 = reinterpret_cast<const float4*>(Bm + (size_t)(kn + bk) * N + n0 + bn);
      const float4* bp1 = reinterpret_cast<const float4*>(Bm + (size_t)(kn + bk + 1) * N + n0 + bn);
      bR[0] = bp0[0]; bR[1] = bp0[1]; bR[2] = bp1[0]; bR[3] = bp1[1];
    }

    v16bf bfr[4];
#pragma unroll
    for (int sn = 0; sn < 4; ++sn) {
      const int nl = wn * 64 + sn * 16 + l16;
      const __bf16* bp = &Bs[buf][nl * BK + hlf * 16];
#pragma unroll
      for (int e = 0; e < 16; ++e) bfr[sn][e] = bp[e];
    }
#pragma unroll
    for (int sm = 0; sm < 2; ++sm) {
      const int ml = wm * 32 + sm * 16 + l16;
      v16bf afrag;
      const __bf16* ap0 = &As[buf][ml * BK + hlf * 16];
      const __bf16* ap1 = &As[buf][ml * BK + hlf * 16 + 8];
#pragma unroll
      for (int e = 0; e < 8; ++e) { afrag[e] = ap0[e]; afrag[e + 8] = ap1[e]; }
#pragma unroll
      for (int sn = 0; sn < 4; ++sn)
        acc[sm][sn] = __builtin_amdgcn_wmma_f32_16x16x32_bf16(
            false, afrag, false, bfr[sn], (short)0, acc[sm][sn], false, false);
    }

    if (more) {
      const int nb = buf ^ 1;
      PK8 p0, p1;
      const float* af = reinterpret_cast<const float*>(aR);
#pragma unroll
      for (int j = 0; j < 8; ++j) { p0.h[j] = (__bf16)af[j]; p1.h[j] = (__bf16)af[8 + j]; }
      *reinterpret_cast<uint4*>(&As[nb][arow * BK + acol])     = p0.u;
      *reinterpret_cast<uint4*>(&As[nb][arow * BK + acol + 8]) = p1.u;
      const float* b0 = reinterpret_cast<const float*>(&bR[0]);
      const float* b1 = reinterpret_cast<const float*>(&bR[2]);
#pragma unroll
      for (int j = 0; j < 8; ++j) {
        PK2 p; p.h[0] = (__bf16)b0[j]; p.h[1] = (__bf16)b1[j];
        *reinterpret_cast<unsigned int*>(&Bs[nb][(bn + j) * BK + bk]) = p.u;
      }
      buf = nb;
    }
  }

  float* cw = C + (size_t)(m0 + wm * 32 + hlf * 8) * N + n0 + wn * 64 + l16;
#pragma unroll
  for (int sm = 0; sm < 2; ++sm)
#pragma unroll
    for (int sn = 0; sn < 4; ++sn)
#pragma unroll
      for (int r = 0; r < 8; ++r)
        cw[(size_t)(sm * 16 + r) * N + sn * 16] = acc[sm][sn][r];
}

// ---------------- WMMA fragment / pack helpers for 64-K matmuls ----------
// A operand: row-major X[64 rows][64 cols], row = output row (two 16B runs).
__device__ __forceinline__ v16bf frag_a64(const __bf16* X, int row, int ks, int hlf) {
  const __bf16* p = X + row * 64 + ks * 32 + hlf * 16;
  v16bf a;
#pragma unroll
  for (int e = 0; e < 8; ++e) { a[e] = p[e]; a[e + 8] = p[8 + e]; }
  return a;
}
// B operand from TRANSPOSED storage Y[N][64 K] (row stride 64): per lane one
// contiguous 32B run -> 2x ds_load_b128.
__device__ __forceinline__ v16bf frag_bT(const __bf16* Y, int col, int ks, int hlf) {
  const __bf16* p = Y + col * 64 + ks * 32 + hlf * 16;
  v16bf bb;
#pragma unroll
  for (int e = 0; e < 16; ++e) bb[e] = p[e];
  return bb;
}
// Pack 8 accumulator lanes -> one 16B bf16 store.
__device__ __forceinline__ void store8_bf16(__bf16* dst, v8f a) {
  PK8 p;
#pragma unroll
  for (int r = 0; r < 8; ++r) p.h[r] = (__bf16)a[r];
  *reinterpret_cast<uint4*>(dst) = p.u;
}
// Load 8 bf16 (16B) -> 8 floats.
__device__ __forceinline__ v8f load8_bf16(const __bf16* src) {
  PK8 p; p.u = *reinterpret_cast<const uint4*>(src);
  v8f o;
#pragma unroll
  for (int r = 0; r < 8; ++r) o[r] = (float)p.h[r];
  return o;
}

// ---------------- Phase A: per-(b,h,chunk) precompute (fully parallel) ----
// Outputs (bf16 unless noted), per unit u = (b*H+h)*NC + ch:
//  kL  [64][64] = Lambda_t * k_t             (rows t)
//  qL  [64][64] = Lambda_t * q_t             (rows t)
//  kRT [64][64] = transposed (Lam_C/Lam_t) k_t   (rows dk)
//  Bi  [64][64] = (I + Akk)^{-1}, Akk[t][j] = beta_t e^{c_t-c_j} (k_t.k_j), j<t
//  Aqk [64][64] = e^{c_t-c_j} (q_t.k_j), j<=t
//  bvT [128][64] = transposed beta_t * v_t   (rows v, cols t)
//  lamC (f32)    = e^{c_63}
__global__ __launch_bounds__(64)
void chunk_prep_kernel(const float* __restrict__ q, const float* __restrict__ k,
                       const float* __restrict__ v, const float* __restrict__ g,
                       const float* __restrict__ beta,
                       __bf16* __restrict__ kL, __bf16* __restrict__ kRT,
                       __bf16* __restrict__ qL, __bf16* __restrict__ Bi_o,
                       __bf16* __restrict__ Aqk_o, __bf16* __restrict__ bvT_o,
                       float* __restrict__ lamC_o) {
  const int u  = blockIdx.x;              // 0..kNU-1
  const int ch = u % kNC;
  const int bh = u / kNC;
  const int b  = bh / kH, hh = bh % kH;
  const int t0 = ch * kCH;
  const int tid = threadIdx.x;            // 0..63 (row t / column c)

  __shared__ float ksh[64][64];
  __shared__ float Ak[64][64];
  __shared__ float Bi[64][64];
  __shared__ float gsh[64], cs[64], bet[64];

  const size_t rowQK = ((size_t)(b * kT + t0 + tid) * kH + hh) * (size_t)kDK;
  float kreg[64], qreg[64];
#pragma unroll
  for (int d = 0; d < 64; ++d) {
    kreg[d] = k[rowQK + d];
    qreg[d] = q[rowQK + d];
    ksh[tid][d] = kreg[d];
  }
  const size_t rgb = (size_t)(b * kT + t0 + tid) * kH + hh;
  gsh[tid] = g[rgb];
  bet[tid] = beta[rgb];
  __syncthreads();

  float c = 0.0f;                         // inclusive prefix of g
  for (int s2 = 0; s2 <= tid; ++s2) c += gsh[s2];
  cs[tid] = c;
  __syncthreads();
  const float clast = cs[63];

  // Akk (LDS) and Aqk (global, bf16); thread owns row i = tid
  for (int j = 0; j < 64; ++j) {
    float dk_ = 0.0f, dq = 0.0f;
#pragma unroll
    for (int d = 0; d < 64; ++d) { dk_ += kreg[d] * ksh[j][d]; dq += qreg[d] * ksh[j][d]; }
    const float r = __expf(c - cs[j]);
    Ak[tid][j] = (j < tid) ? bet[tid] * r * dk_ : 0.0f;
    Aqk_o[(size_t)u * 4096 + tid * 64 + j] = (__bf16)((j <= tid) ? r * dq : 0.0f);
  }
  __syncthreads();

  // forward substitution: Bi = (I+Ak)^{-1}; thread owns COLUMN c = tid.
  // Bi[j][tid] is private to this thread -> no barriers needed inside.
  for (int i = 0; i < 64; ++i) {
    float s = (tid == i) ? 1.0f : 0.0f;
    for (int j = 0; j < i; ++j) s -= Ak[i][j] * Bi[j][tid];
    Bi[i][tid] = s;
  }
  __syncthreads();

  for (int cc = 0; cc < 64; ++cc)
    Bi_o[(size_t)u * 4096 + tid * 64 + cc] = (__bf16)Bi[tid][cc];

  const float lam = __expf(c);
  const float rr  = __expf(clast - c);
#pragma unroll
  for (int d = 0; d < 64; ++d) {
    kL [(size_t)u * 4096 + tid * 64 + d] = (__bf16)(kreg[d] * lam);
    qL [(size_t)u * 4096 + tid * 64 + d] = (__bf16)(qreg[d] * lam);
    kRT[(size_t)u * 4096 + d * 64 + tid] = (__bf16)(kreg[d] * rr);
  }
  const size_t rv = ((size_t)(b * kT + t0 + tid) * kH + hh) * (size_t)kDV;
  for (int vv = 0; vv < kDV; ++vv)
    bvT_o[(size_t)u * 8192 + vv * 64 + tid] = (__bf16)(bet[tid] * v[rv + vv]);
  if (tid == 0) lamC_o[u] = __expf(clast);
}

// ---------------- Phase B: chunked scan, 16 sequential WMMA steps per head --
// One block per (b,h); 256 threads = 8 waves. State S[64dk][128dv] lives in
// WMMA accumulators (8 waves x 4 tiles of 16x16). All LDS intermediates are
// stored TRANSPOSED [v][k] so B-fragment reads are 2x ds_load_b128 and
// epilogue writes are 1x packed ds_store_b128 per tile.
__global__ __launch_bounds__(256)
void chunk_scan_kernel(const __bf16* __restrict__ kL, const __bf16* __restrict__ kRT,
                       const __bf16* __restrict__ qL, const __bf16* __restrict__ Bi,
                       const __bf16* __restrict__ Aqk, const __bf16* __restrict__ bvT,
                       const float* __restrict__ lamC, const float* __restrict__ beta,
                       float* __restrict__ o) {
  const int bh  = blockIdx.x;
  const int b   = bh / kH, hh = bh % kH;
  const int tid = threadIdx.x, lane = tid & 31, w = tid >> 5;
  const int mt  = w & 3;          // row-tile (rows mt*16..mt*16+15)
  const int nh  = w >> 2;         // 0..1 -> col tiles nh*4 .. nh*4+3
  const int l16 = lane & 15, hlf = lane >> 4;
  const int rowb = mt * 16 + hlf * 8;        // C-layout local row base

  __shared__ __align__(16) __bf16 S0T[128 * 64];  // [v][dk]
  __shared__ __align__(16) __bf16 b0T[128 * 64];  // [v][t]
  __shared__ __align__(16) __bf16 UT [128 * 64];  // [v][t]
  __shared__ float bet_s[64];

  v8f accS[4];
#pragma unroll
  for (int i = 0; i < 4; ++i)
#pragma unroll
    for (int r = 0; r < 8; ++r) accS[i][r] = 0.0f;

  for (int ch = 0; ch < kNC; ++ch) {
    const size_t u  = (size_t)bh * kNC + ch;
    const int    t0 = ch * kCH;
    const __bf16* kL_g  = kL  + u * 4096;
    const __bf16* kRT_g = kRT + u * 4096;
    const __bf16* qL_g  = qL  + u * 4096;
    const __bf16* Bi_g  = Bi  + u * 4096;
    const __bf16* Aq_g  = Aqk + u * 4096;
    const __bf16* bvT_g = bvT + u * 8192;
    const float   lc    = lamC[u];

    // 1) publish S (bf16, transposed) and beta chunk
    if (tid < 64) bet_s[tid] = beta[(size_t)(b * kT + t0 + tid) * kH + hh];
#pragma unroll
    for (int i = 0; i < 4; ++i) {
      const int v0 = (nh * 4 + i) * 16 + l16;
      store8_bf16(&S0T[v0 * 64 + rowb], accS[i]);
    }
    __syncthreads();

    // 2) W = kL @ S0 ; b0 = bv - beta .* W   (stored transposed)
    v8f acc[4];
#pragma unroll
    for (int i = 0; i < 4; ++i)
#pragma unroll
      for (int r = 0; r < 8; ++r) acc[i][r] = 0.0f;
#pragma unroll
    for (int ks = 0; ks < 2; ++ks) {
      const v16bf af = frag_a64(kL_g, mt * 16 + l16, ks, hlf);
#pragma unroll
      for (int i = 0; i < 4; ++i)
        acc[i] = __builtin_amdgcn_wmma_f32_16x16x32_bf16(
            false, af, false, frag_bT(S0T, (nh * 4 + i) * 16 + l16, ks, hlf),
            (short)0, acc[i], false, false);
    }
#pragma unroll
    for (int i = 0; i < 4; ++i) {
      const int v0 = (nh * 4 + i) * 16 + l16;
      const v8f bvv = load8_bf16(bvT_g + v0 * 64 + rowb);
      v8f t;
#pragma unroll
      for (int r = 0; r < 8; ++r) t[r] = bvv[r] - bet_s[rowb + r] * acc[i][r];
      store8_bf16(&b0T[v0 * 64 + rowb], t);
    }
    __syncthreads();

    // 3) U = Binv @ b0
#pragma unroll
    for (int i = 0; i < 4; ++i)
#pragma unroll
      for (int r = 0; r < 8; ++r) acc[i][r] = 0.0f;
#pragma unroll
    for (int ks = 0; ks < 2; ++ks) {
      const v16bf af = frag_a64(Bi_g, mt * 16 + l16, ks, hlf);
#pragma unroll
      for (int i = 0; i < 4; ++i)
        acc[i] = __builtin_amdgcn_wmma_f32_16x16x32_bf16(
            false, af, false, frag_bT(b0T, (nh * 4 + i) * 16 + l16, ks, hlf),
            (short)0, acc[i], false, false);
    }
#pragma unroll
    for (int i = 0; i < 4; ++i) {
      const int v0 = (nh * 4 + i) * 16 + l16;
      store8_bf16(&UT[v0 * 64 + rowb], acc[i]);
    }
    __syncthreads();

    // 4) o = Aqk @ U + qL @ S0
#pragma unroll
    for (int i = 0; i < 4; ++i)
#pragma unroll
      for (int r = 0; r < 8; ++r) acc[i][r] = 0.0f;
#pragma unroll
    for (int ks = 0; ks < 2; ++ks) {
      const v16bf af = frag_a64(Aq_g, mt * 16 + l16, ks, hlf);
#pragma unroll
      for (int i = 0; i < 4; ++i)
        acc[i] = __builtin_amdgcn_wmma_f32_16x16x32_bf16(
            false, af, false, frag_bT(UT, (nh * 4 + i) * 16 + l16, ks, hlf),
            (short)0, acc[i], false, false);
    }
#pragma unroll
    for (int ks = 0; ks < 2; ++ks) {
      const v16bf af = frag_a64(qL_g, mt * 16 + l16, ks, hlf);
#pragma unroll
      for (int i = 0; i < 4; ++i)
        acc[i] = __builtin_amdgcn_wmma_f32_16x16x32_bf16(
            false, af, false, frag_bT(S0T, (nh * 4 + i) * 16 + l16, ks, hlf),
            (short)0, acc[i], false, false);
    }
#pragma unroll
    for (int i = 0; i < 4; ++i) {
      const int v0 = (nh * 4 + i) * 16 + l16;
#pragma unroll
      for (int r = 0; r < 8; ++r)
        o[((size_t)(b * kT + t0 + rowb + r) * kH + hh) * kDV + v0] = acc[i][r];
    }

    // 5) S = lamC * S + kRT @ U   (decay in registers, update via WMMA)
#pragma unroll
    for (int i = 0; i < 4; ++i)
#pragma unroll
      for (int r = 0; r < 8; ++r) accS[i][r] *= lc;
#pragma unroll
    for (int ks = 0; ks < 2; ++ks) {
      const v16bf af = frag_a64(kRT_g, mt * 16 + l16, ks, hlf);
#pragma unroll
      for (int i = 0; i < 4; ++i)
        accS[i] = __builtin_amdgcn_wmma_f32_16x16x32_bf16(
            false, af, false, frag_bT(UT, (nh * 4 + i) * 16 + l16, ks, hlf),
            (short)0, accS[i], false, false);
    }
    __syncthreads();   // protect S0T/b0T/UT before next chunk rewrites them
  }
}

// ---------------- RMSNorm (one block per row) ----------------
__global__ void rmsnorm_kernel(const float* __restrict__ x, const float* __restrict__ w,
                               float* __restrict__ out, int D, float eps) {
  const int row = blockIdx.x;
  const float* xr = x + (size_t)row * D;
  float s = 0.0f;
  for (int i = threadIdx.x; i < D; i += blockDim.x) { float v = xr[i]; s += v * v; }
  __shared__ float red[256];
  red[threadIdx.x] = s;
  __syncthreads();
  for (int off = 128; off > 0; off >>= 1) {
    if ((int)threadIdx.x < off) red[threadIdx.x] += red[threadIdx.x + off];
    __syncthreads();
  }
  const float r = rsqrtf(red[0] / (float)D + eps);
  for (int i = threadIdx.x; i < D; i += blockDim.x)
    out[(size_t)row * D + i] = xr[i] * r * w[i];
}

// ---------------- a/b projections -> decay g and beta ----------------
__global__ void ab_proj_kernel(const float* __restrict__ h,
                               const float* __restrict__ Wa, const float* __restrict__ Wb,
                               const float* __restrict__ dt_bias, const float* __restrict__ A_log,
                               float* __restrict__ g, float* __restrict__ beta) {
  const int idx = blockIdx.x * blockDim.x + threadIdx.x;
  if (idx >= kNT * kH) return;
  const int row = idx / kH, hh = idx % kH;
  const float* hr = h + (size_t)row * kD;
  float a = 0.0f, b = 0.0f;
  for (int d = 0; d < kD; ++d) {
    const float hv = hr[d];
    a += hv * Wa[d * kH + hh];
    b += hv * Wb[d * kH + hh];
  }
  a += dt_bias[hh];
  const float sp = (a > 20.0f) ? a : log1pf(__expf(a));        // softplus
  g[idx]    = -__expf(A_log[hh]) * sp;
  beta[idx] = 1.0f / (1.0f + __expf(-b));
}

// ---------------- causal depthwise conv (K=4) + SiLU ----------------
__global__ void conv_silu_kernel(const float* __restrict__ in, const float* __restrict__ w,
                                 float* __restrict__ out, int C) {
  const size_t idx = (size_t)blockIdx.x * blockDim.x + threadIdx.x;
  const size_t total = (size_t)kB * kT * C;
  if (idx >= total) return;
  const int c = (int)(idx % C);
  const int t = (int)((idx / C) % kT);
  const int b = (int)(idx / ((size_t)C * kT));
  float acc = 0.0f;
#pragma unroll
  for (int j = 0; j < kCONV; ++j) {
    const int tt = t - (kCONV - 1) + j;
    if (tt >= 0) acc += in[((size_t)(b * kT + tt)) * C + c] * w[c * kCONV + j];
  }
  out[idx] = acc / (1.0f + __expf(-acc));
}

// ---------------- per-head l2norm over DK (in place) ----------------
__global__ void l2norm_kernel(float* __restrict__ x, int nrows, float scale) {
  const int r = blockIdx.x * blockDim.x + threadIdx.x;
  if (r >= nrows) return;
  float* xr = x + (size_t)r * kDK;
  float s = 0.0f;
#pragma unroll 8
  for (int i = 0; i < kDK; ++i) { const float v = xr[i]; s += v * v; }
  const float rr = rsqrtf(s + 1e-6f) * scale;
#pragma unroll 8
  for (int i = 0; i < kDK; ++i) xr[i] *= rr;
}

// ---------------- gated RMSNorm * SiLU(gate) ----------------
__global__ void gate_norm_kernel(const float* __restrict__ o, const float* __restrict__ gate,
                                 const float* __restrict__ w, float* __restrict__ out) {
  const int row = blockIdx.x * blockDim.x + threadIdx.x;     // B*T*H rows
  if (row >= kNT * kH) return;
  const float* orow = o    + (size_t)row * kDV;
  const float* grow = gate + (size_t)row * kDV;
  float s = 0.0f;
#pragma unroll 8
  for (int i = 0; i < kDV; ++i) { const float x = orow[i]; s += x * x; }
  const float r = rsqrtf(s / (float)kDV + 1e-5f);
#pragma unroll 4
  for (int i = 0; i < kDV; ++i) {
    const float gv  = grow[i];
    const float sig = gv / (1.0f + __expf(-gv));
    out[(size_t)row * kDV + i] = orow[i] * r * w[i] * sig;
  }
}

// ---------------- elementwise helpers ----------------
__global__ void add_kernel(const float* __restrict__ a, const float* __restrict__ b,
                           float* __restrict__ out, size_t n) {
  const size_t i = (size_t)blockIdx.x * blockDim.x + threadIdx.x;
  if (i < n) out[i] = a[i] + b[i];
}

__global__ void silu_mul_kernel(float* __restrict__ t1, const float* __restrict__ t3, size_t n) {
  const size_t i = (size_t)blockIdx.x * blockDim.x + threadIdx.x;
  if (i < n) {
    const float x = t1[i];
    t1[i] = (x / (1.0f + __expf(-x))) * t3[i];
  }
}

// ---------------- launch ----------------
extern "C" void kernel_launch(void* const* d_in, const int* in_sizes, int n_in,
                              void* d_out, int out_size, void* d_ws, size_t ws_size,
                              hipStream_t stream) {
  (void)in_sizes; (void)n_in; (void)out_size; (void)ws_size;

  const float* x        = (const float*)d_in[0];
  const float* norm1_w  = (const float*)d_in[1];
  const float* norm2_w  = (const float*)d_in[2];
  const float* Wq       = (const float*)d_in[3];
  const float* Wk       = (const float*)d_in[4];
  const float* Wv       = (const float*)d_in[5];
  const float* conv_q   = (const float*)d_in[6];
  const float* conv_k   = (const float*)d_in[7];
  const float* conv_v   = (const float*)d_in[8];
  const float* Wa       = (const float*)d_in[9];
  const float* Wb       = (const float*)d_in[10];
  const float* dt_bias  = (const float*)d_in[11];
  const float* A_log    = (const float*)d_in[12];
  const float* Wg       = (const float*)d_in[13];
  const float* o_norm_w = (const float*)d_in[14];
  const float* Wo       = (const float*)d_in[15];
  const float* W1       = (const float*)d_in[16];
  const float* W3       = (const float*)d_in[17];
  const float* W2       = (const float*)d_in[18];
  float* out = (float*)d_out;

  // workspace layout (floats); aliased buffers noted
  float* ws = (float*)d_ws;
  constexpr size_t SZ = (size_t)kNT * kD;              // 2M floats
  float* h    = ws + 0;                                // [NT, D]   (reused as h2)
  float* hq   = ws + 1 * SZ;                           // [NT, H*DK]
  float* hk   = ws + 2 * SZ;                           // [NT, H*DK]
  float* hv   = ws + 3 * SZ;                           // [NT, H*DV] (2*SZ; reused as og)
  float* hg   = ws + 5 * SZ;                           // [NT, H*DV] (2*SZ)
  float* q    = ws + 7 * SZ;                           // [NT, H*DK]
  float* k    = ws + 8 * SZ;                           // [NT, H*DK]
  float* v    = ws + 9 * SZ;                           // [NT, H*DV] (2*SZ)
  float* o    = ws + 11 * SZ;                          // [NT, H*DV] (2*SZ)
  float* x1   = ws + 13 * SZ;                          // [NT, D]
  float* tmp  = ws + 14 * SZ;                          // [NT, D]
  float* ffn  = ws + 15 * SZ;                          // [NT, D]
  float* gdec = ws + 16 * SZ;                          // [NT, H]
  float* beta = ws + 16 * SZ + (size_t)kNT * kH;       // [NT, H]
  // chunked-scan precompute block (bf16), starts at 17*SZ
  __bf16* kLb  = (__bf16*)(ws + 17 * SZ);              // [512][4096]
  __bf16* kRTb = kLb  + (size_t)kNU * 4096;
  __bf16* qLb  = kRTb + (size_t)kNU * 4096;
  __bf16* Bib  = qLb  + (size_t)kNU * 4096;
  __bf16* Aqb  = Bib  + (size_t)kNU * 4096;
  __bf16* bvTb = Aqb  + (size_t)kNU * 4096;            // [512][8192] (transposed)
  float*  lamC = (float*)(bvTb + (size_t)kNU * 8192);  // [512]
  float* og   = hv;          // gated output, alias (hv dead after conv v)
  float* t1   = hq;          // [NT, FH] spans hq..hv (those buffers dead by FFN time)
  float* t3   = hg;          // [NT, FH] spans hg..q (dead by FFN time)
  float* h2   = h;

  const dim3 blk256(256);
  const int CQ = kH * kDK;   // 1024
  const int CV = kH * kDV;   // 2048

  // 1) pre-norm
  rmsnorm_kernel<<<kNT, blk256, 0, stream>>>(x, norm1_w, h, kD, 1e-6f);

  // 2) projections (WMMA GEMMs)
  wmma_gemm_kernel<1024><<<dim3(CQ / BN, kNT / BM), blk256, 0, stream>>>(h, Wq, hq, kNT, kD);
  wmma_gemm_kernel<1024><<<dim3(CQ / BN, kNT / BM), blk256, 0, stream>>>(h, Wk, hk, kNT, kD);
  wmma_gemm_kernel<2048><<<dim3(CV / BN, kNT / BM), blk256, 0, stream>>>(h, Wv, hv, kNT, kD);
  wmma_gemm_kernel<2048><<<dim3(CV / BN, kNT / BM), blk256, 0, stream>>>(h, Wg, hg, kNT, kD);

  // 3) decay + beta
  ab_proj_kernel<<<(kNT * kH + 255) / 256, blk256, 0, stream>>>(h, Wa, Wb, dt_bias, A_log, gdec, beta);

  // 4) short conv + silu, then l2norm(q)*scale, l2norm(k)
  conv_silu_kernel<<<(int)(((size_t)kB * kT * CQ + 255) / 256), blk256, 0, stream>>>(hq, conv_q, q, CQ);
  conv_silu_kernel<<<(int)(((size_t)kB * kT * CQ + 255) / 256), blk256, 0, stream>>>(hk, conv_k, k, CQ);
  conv_silu_kernel<<<(int)(((size_t)kB * kT * CV + 255) / 256), blk256, 0, stream>>>(hv, conv_v, v, CV);
  l2norm_kernel<<<(kNT * kH + 255) / 256, blk256, 0, stream>>>(q, kNT * kH, 0.125f); // DK^-0.5
  l2norm_kernel<<<(kNT * kH + 255) / 256, blk256, 0, stream>>>(k, kNT * kH, 1.0f);

  // 5) chunked gated delta rule: parallel precompute + WMMA scan
  chunk_prep_kernel<<<kNU, 64, 0, stream>>>(q, k, v, gdec, beta,
                                            kLb, kRTb, qLb, Bib, Aqb, bvTb, lamC);
  chunk_scan_kernel<<<kB * kH, blk256, 0, stream>>>(kLb, kRTb, qLb, Bib, Aqb, bvTb,
                                                    lamC, beta, o);

  // 6) gated RMSNorm * SiLU(gate); output projection; residual 1
  gate_norm_kernel<<<(kNT * kH + 255) / 256, blk256, 0, stream>>>(o, hg, o_norm_w, og);
  wmma_gemm_kernel<1024><<<dim3(kD / BN, kNT / BM), blk256, 0, stream>>>(og, Wo, tmp, kNT, CV);
  add_kernel<<<(int)((SZ + 255) / 256), blk256, 0, stream>>>(x, tmp, x1, SZ);

  // 7) FFN: pre-norm, W1/W3, silu-mul, W2, residual 2
  rmsnorm_kernel<<<kNT, blk256, 0, stream>>>(x1, norm2_w, h2, kD, 1e-6f);
  wmma_gemm_kernel<2816><<<dim3(kFH / BN, kNT / BM), blk256, 0, stream>>>(h2, W1, t1, kNT, kD);
  wmma_gemm_kernel<2816><<<dim3(kFH / BN, kNT / BM), blk256, 0, stream>>>(h2, W3, t3, kNT, kD);
  silu_mul_kernel<<<(int)(((size_t)kNT * kFH + 255) / 256), blk256, 0, stream>>>(t1, t3, (size_t)kNT * kFH);
  wmma_gemm_kernel<1024><<<dim3(kD / BN, kNT / BM), blk256, 0, stream>>>(t1, W2, ffn, kNT, kFH);
  add_kernel<<<(int)((SZ + 255) / 256), blk256, 0, stream>>>(x1, ffn, out, SZ);
}
